// SwinMixerBlock_27513560498657
// MI455X (gfx1250) — matrix operand order repaired and checked by reading, not compiled, not run on hardware
//
#include <hip/hip_runtime.h>
#include <hip/hip_bf16.h>

// ---------------------------------------------------------------------------
// Swin mixer block for MI455X (gfx1250, wave32, WMMA bf16 16x16x32 + TDM)
// ---------------------------------------------------------------------------
// B=32 SEQ=900 GRID=30 PFX=16 D=512 H=8 HD=64 WS=3 SS=1 NW=100 N=9 INTER=1536

typedef __bf16 bf16_t;
typedef __attribute__((ext_vector_type(16))) __bf16 v16bf;
typedef __attribute__((ext_vector_type(8)))  float  v8f;
typedef __attribute__((ext_vector_type(4)))  unsigned int v4u;
typedef __attribute__((ext_vector_type(8)))  int v8i_t;
typedef __attribute__((ext_vector_type(4)))  int v4i_t;

#define Bc    32
#define SEQc  900
#define GRIDc 30
#define PFXc  16
#define Dc    512
#define Hc    8
#define HDc   64
#define NWIN  3200           // B * NW
#define NTOK  28800          // NWIN * 9
#define INTERc 1536
#define TOTROWS (Bc*916)     // 29312
#define EPSc  1e-5f

#if __has_builtin(__builtin_amdgcn_tensor_load_to_lds)
#define USE_TDM 1
#else
#define USE_TDM 0
#endif

// ---------------------------------------------------------------------------
// Tensor Data Mover: DMA a rows x 32 bf16 tile (row stride = Kelem elements)
// from global memory into LDS.  Descriptor per CDNA5 ISA chapter 8:
//   group0: count=1 | lds_addr | global_addr(57b) | type=2
//   group1: data_size=1(2B), tensor_dim0=Kelem, tensor_dim1=big,
//           tile_dim0=32, tile_dim1=rows, tensor_dim0_stride=Kelem
//   group2/3: zero (2-D tensor)
// ---------------------------------------------------------------------------
__device__ __forceinline__ void tdm_load_tile(const bf16_t* gsrc,
                                              unsigned lds_addr, int rows,
                                              int Kelem) {
#if USE_TDM
  unsigned long long ga = (unsigned long long)(size_t)gsrc;
  v4u g0;
  g0[0] = 1u;                                        // count=1 (valid D#)
  g0[1] = lds_addr;                                  // LDS byte address
  g0[2] = (unsigned)ga;                              // global addr [31:0]
  g0[3] = (unsigned)((ga >> 32) & 0x01ffffffu)       // global addr [56:32]
          | (2u << 30);                              // type = 2 ("image")
  unsigned td0 = (unsigned)Kelem;                    // tensor dim0 (elems)
  unsigned td1 = 1u << 20;                           // tensor dim1 (large)
  v8i_t g1;
  g1[0] = (int)(1u << 16);                           // data_size = 1 (2 bytes)
  g1[1] = (int)((td0 & 0xffffu) << 16);              // abar=0 | td0 lo16
  g1[2] = (int)((td0 >> 16) | ((td1 & 0xffffu) << 16));
  g1[3] = (int)((td1 >> 16) | (32u << 16));          // tile_dim0 = 32
  g1[4] = (int)(unsigned)rows;                       // tile_dim1, tile_dim2=0
  g1[5] = (int)(unsigned)Kelem;                      // dim0 stride [31:0]
  g1[6] = 0;                                         // stride hi / dim1 stride
  g1[7] = 0;
  v4i_t gz = {0, 0, 0, 0};
#if __clang_major__ >= 23
  v8i_t gz8 = {0, 0, 0, 0, 0, 0, 0, 0};
  __builtin_amdgcn_tensor_load_to_lds(g0, g1, gz, gz, gz8, 0);
#else
  __builtin_amdgcn_tensor_load_to_lds(g0, g1, gz, gz, 0);
#endif
#endif
}

__device__ __forceinline__ void tdm_wait() {
#if USE_TDM
  __builtin_amdgcn_s_wait_tensorcnt(0);
#endif
}

#define LDSADDR(arr) ((unsigned)(size_t)(&(arr)[0]))

// -------- fallback manual staging (global_load_b128 + ds_store_b128) -------
__device__ __forceinline__ void man_tileA(const bf16_t* __restrict__ A,
                                          bf16_t* ldsA, long mBase, int K,
                                          int k0, int tid) {
  int lin = tid * 8;
  {
    int row = lin >> 5, col = lin & 31;
    *(uint4*)(ldsA + lin) =
        *(const uint4*)(A + (size_t)(mBase + row) * K + k0 + col);
  }
  lin += 2048;
  {
    int row = lin >> 5, col = lin & 31;
    *(uint4*)(ldsA + lin) =
        *(const uint4*)(A + (size_t)(mBase + row) * K + k0 + col);
  }
}
__device__ __forceinline__ void man_tileB(const bf16_t* __restrict__ W,
                                          bf16_t* ldsB, long nBase, int K,
                                          int k0, int tid) {
  int lin = tid * 8;
  int row = lin >> 5, col = lin & 31;
  *(uint4*)(ldsB + lin) =
      *(const uint4*)(W + (size_t)(nBase + row) * K + k0 + col);
}

// stage one (A,B) tile pair into the given LDS buffers
__device__ __forceinline__ void stage2(const bf16_t* A, const bf16_t* W,
                                       long mBase, long nBase, int K, int k0,
                                       int tid, int wave, bf16_t* dA,
                                       bf16_t* dB) {
#if USE_TDM
  if (wave == 0) {
    tdm_load_tile(A + (size_t)mBase * K + k0, LDSADDR(dA), 128, K);
    tdm_load_tile(W + (size_t)nBase * K + k0, LDSADDR(dB), 64, K);
  }
#else
  man_tileA(A, dA, mBase, K, k0, tid);
  man_tileB(W, dB, nBase, K, k0, tid);
#endif
}
// stage (A, Bgate, Bup)
__device__ __forceinline__ void stage3(const bf16_t* A, const bf16_t* W,
                                       long mBase, long nBase, int K, int k0,
                                       int tid, int wave, bf16_t* dA,
                                       bf16_t* dB, bf16_t* dB2) {
#if USE_TDM
  if (wave == 0) {
    tdm_load_tile(A + (size_t)mBase * K + k0, LDSADDR(dA), 128, K);
    tdm_load_tile(W + (size_t)nBase * K + k0, LDSADDR(dB), 64, K);
    tdm_load_tile(W + (size_t)(nBase + INTERc) * K + k0, LDSADDR(dB2), 64, K);
  }
#else
  man_tileA(A, dA, mBase, K, k0, tid);
  man_tileB(W, dB, nBase, K, k0, tid);
  man_tileB(W, dB2, nBase + INTERc, K, k0, tid);
#endif
}

// ---------------------------------------------------------------------------
// WMMA fragment gathers from LDS (bf16, 16x16x32 layouts per ISA 7.12.2)
// ---------------------------------------------------------------------------
__device__ __forceinline__ v16bf fragA(const bf16_t* ldsA, int m0, int lane) {
  v16bf a;
  int mm = m0 + (lane & 15);
  int kb = (lane >> 4) * 8;
#pragma unroll
  for (int e = 0; e < 16; e++) {
    int k = ((e >> 3) << 4) + kb + (e & 7);
    a[e] = ldsA[mm * 32 + k];
  }
  return a;
}
__device__ __forceinline__ v16bf fragB(const bf16_t* ldsB, int n0, int lane) {
  v16bf b;
  int nn = n0 + (lane & 15);
  int kb = (lane >> 4) * 8;
#pragma unroll
  for (int e = 0; e < 16; e++) b[e] = ldsB[nn * 32 + ((e >> 3) << 4) + kb + (e & 7)];
  return b;
}

// Workgroup tile = 128(M) x 64(N); 8 waves, wave covers 64x16 (4 WMMA tiles).
// Double-buffered LDS; TDM prefetches next K-slab during compute.
#define GEMM_PROLOGUE()                                                        \
  __shared__ __align__(16) bf16_t ldsA[2][128 * 32];                           \
  __shared__ __align__(16) bf16_t ldsB[2][64 * 32];                            \
  const int tid = threadIdx.x, lane = tid & 31, wave = tid >> 5;               \
  const int wm = wave & 1, wn = wave >> 1;                                     \
  const long mBase = (long)blockIdx.x * 128;                                   \
  const long nBase = (long)blockIdx.y * 64;

#define GEMM_MAINLOOP(Aptr, Wptr, Kdim)                                        \
  v8f acc[4] = {};                                                             \
  stage2((Aptr), (Wptr), mBase, nBase, (Kdim), 0, tid, wave, ldsA[0],          \
         ldsB[0]);                                                             \
  for (int k0 = 0; k0 < (Kdim); k0 += 32) {                                    \
    const int cur = (k0 >> 5) & 1;                                             \
    tdm_wait();                                                                \
    __syncthreads();                                                           \
    if (k0 + 32 < (Kdim))                                                      \
      stage2((Aptr), (Wptr), mBase, nBase, (Kdim), k0 + 32, tid, wave,         \
             ldsA[cur ^ 1], ldsB[cur ^ 1]);                                    \
    v16bf bfr = fragB(ldsB[cur], wn * 16, lane);                               \
    _Pragma("unroll") for (int t = 0; t < 4; t++) {                            \
      v16bf afr = fragA(ldsA[cur], wm * 64 + t * 16, lane);                    \
      acc[t] = __builtin_amdgcn_wmma_f32_16x16x32_bf16(                        \
          false, afr, false, bfr, (short)0, acc[t], false, false);             \
    }                                                                          \
    __syncthreads();                                                           \
  }

// ------------------------- qkv GEMM: out bf16 + bias -----------------------
__global__ void __launch_bounds__(256)
gemm_qkv_kernel(const bf16_t* __restrict__ A, const bf16_t* __restrict__ W,
                const float* __restrict__ bias, bf16_t* __restrict__ out) {
  GEMM_PROLOGUE();
  GEMM_MAINLOOP(A, W, Dc);
  const int n = (int)nBase + wn * 16 + (lane & 15);
  const float bn = bias[n];
#pragma unroll
  for (int t = 0; t < 4; t++) {
    int mrow = (int)mBase + wm * 64 + t * 16 + ((lane >> 4) << 3);
#pragma unroll
    for (int r = 0; r < 8; r++)
      out[(size_t)(mrow + r) * (3 * Dc) + n] = (bf16_t)(acc[t][r] + bn);
  }
}

// ------------- proj GEMM: bias + un-window + un-roll + residual ------------
__global__ void __launch_bounds__(256)
gemm_proj_kernel(const bf16_t* __restrict__ A, const bf16_t* __restrict__ W,
                 const float* __restrict__ bias,
                 const float* __restrict__ residual,
                 float* __restrict__ hs_pre) {
  GEMM_PROLOGUE();
  GEMM_MAINLOOP(A, W, Dc);
  const int n = (int)nBase + wn * 16 + (lane & 15);
  const float bn = bias[n];
#pragma unroll
  for (int t = 0; t < 4; t++) {
    int mrow = (int)mBase + wm * 64 + t * 16 + ((lane >> 4) << 3);
#pragma unroll
    for (int r = 0; r < 8; r++) {
      int m = mrow + r;
      int w = m / 9, i = m - w * 9;
      int b = w / 100, wr = w - b * 100;
      int wi = wr / 10, wj = wr - wi * 10;
      int hp = wi * 3 + i / 3, wp = wj * 3 + i % 3;
      int h = (hp + 1) % GRIDc, ww = (wp + 1) % GRIDc;
      int pos = h * GRIDc + ww;
      float val = acc[t][r] + bn +
                  residual[((size_t)b * SEQc + pos) * Dc + n];
      hs_pre[((size_t)b * 916 + PFXc + pos) * Dc + n] = val;
    }
  }
}

// --------------- fused gate/up GEMM + SiLU(g)*u, out bf16 ------------------
__global__ void __launch_bounds__(256)
gemm_gu_kernel(const bf16_t* __restrict__ A, const bf16_t* __restrict__ W,
               bf16_t* __restrict__ out, int K) {
  __shared__ __align__(16) bf16_t ldsA[2][128 * 32];
  __shared__ __align__(16) bf16_t ldsB[2][64 * 32];
  __shared__ __align__(16) bf16_t ldsB2[2][64 * 32];
  const int tid = threadIdx.x, lane = tid & 31, wave = tid >> 5;
  const int wm = wave & 1, wn = wave >> 1;
  const long mBase = (long)blockIdx.x * 128;
  const long nBase = (long)blockIdx.y * 64;
  v8f accg[4] = {}, accu[4] = {};
  stage3(A, W, mBase, nBase, K, 0, tid, wave, ldsA[0], ldsB[0], ldsB2[0]);
  for (int k0 = 0; k0 < K; k0 += 32) {
    const int cur = (k0 >> 5) & 1;
    tdm_wait();
    __syncthreads();
    if (k0 + 32 < K)
      stage3(A, W, mBase, nBase, K, k0 + 32, tid, wave, ldsA[cur ^ 1],
             ldsB[cur ^ 1], ldsB2[cur ^ 1]);
    v16bf bg = fragB(ldsB[cur], wn * 16, lane);
    v16bf bu = fragB(ldsB2[cur], wn * 16, lane);
#pragma unroll
    for (int t = 0; t < 4; t++) {
      v16bf afr = fragA(ldsA[cur], wm * 64 + t * 16, lane);
      accg[t] = __builtin_amdgcn_wmma_f32_16x16x32_bf16(
          false, afr, false, bg, (short)0, accg[t], false, false);
      accu[t] = __builtin_amdgcn_wmma_f32_16x16x32_bf16(
          false, afr, false, bu, (short)0, accu[t], false, false);
    }
    __syncthreads();
  }
  const int n = (int)nBase + wn * 16 + (lane & 15);
#pragma unroll
  for (int t = 0; t < 4; t++) {
    int mrow = (int)mBase + wm * 64 + t * 16 + ((lane >> 4) << 3);
#pragma unroll
    for (int r = 0; r < 8; r++) {
      float g = accg[t][r], u = accu[t][r];
      float s = g / (1.f + __expf(-g));
      out[(size_t)(mrow + r) * INTERc + n] = (bf16_t)(s * u);
    }
  }
}

// ------------- down GEMM: + addend, fp32 out; mode1 = prefix rows ----------
__global__ void __launch_bounds__(256)
gemm_down_kernel(const bf16_t* __restrict__ A, const bf16_t* __restrict__ W,
                 const float* __restrict__ addend, float* __restrict__ dst,
                 int K, int mode) {
  GEMM_PROLOGUE();
  GEMM_MAINLOOP(A, W, K);
  const int n = (int)nBase + wn * 16 + (lane & 15);
#pragma unroll
  for (int t = 0; t < 4; t++) {
    int mrow = (int)mBase + wm * 64 + t * 16 + ((lane >> 4) << 3);
#pragma unroll
    for (int r = 0; r < 8; r++) {
      int m = mrow + r;
      size_t idx = mode ? ((size_t)(m >> 4) * 916 + (m & 15)) : (size_t)m;
      dst[idx * Dc + n] = acc[t][r] + addend[idx * Dc + n];
    }
  }
}

// ------------------- windowed 9x9 attention (one block/window) -------------
__global__ void __launch_bounds__(256)
attn_kernel(const bf16_t* __restrict__ qkv, const float* __restrict__ rel_bias,
            bf16_t* __restrict__ out) {
  __shared__ float sq[Hc][576];
  __shared__ float sk[Hc][576];
  __shared__ float sv[Hc][576];
  __shared__ float sp[Hc][81];
  const int tid = threadIdx.x, lane = tid & 31, h = tid >> 5;
  const int w = blockIdx.x;                // window id 0..3199
  const int wr = w % 100, wi = wr / 10, wj = wr % 10;
  const size_t base = (size_t)w * 9 * (3 * Dc);
  for (int idx = lane; idx < 576; idx += 32) {
    int i = idx >> 6, d = idx & 63;
    size_t r = base + (size_t)i * (3 * Dc) + h * HDc + d;
    sq[h][idx] = (float)qkv[r] * 0.125f;   // HD^-0.5
    sk[h][idx] = (float)qkv[r + Dc];
    sv[h][idx] = (float)qkv[r + 2 * Dc];
  }
  __syncthreads();
  for (int p = lane; p < 81; p += 32) {
    int i = p / 9, j = p - i * 9;
    float dot = 0.f;
#pragma unroll
    for (int d = 0; d < 64; d++) dot += sq[h][i * 64 + d] * sk[h][j * 64 + d];
    int dy = i / 3 - j / 3, dx = i % 3 - j % 3;
    float bias = rel_bias[((dy + 2) * 5 + (dx + 2)) * Hc + h];
    int hi = wi * 3 + i / 3, wwi = wj * 3 + i % 3;
    int hj = wi * 3 + j / 3, wwj = wj * 3 + j % 3;
    int li = (hi < 27 ? 0 : (hi < 29 ? 1 : 2)) * 3 +
             (wwi < 27 ? 0 : (wwi < 29 ? 1 : 2));
    int lj = (hj < 27 ? 0 : (hj < 29 ? 1 : 2)) * 3 +
             (wwj < 27 ? 0 : (wwj < 29 ? 1 : 2));
    sp[h][p] = dot + bias + ((li != lj) ? -100.f : 0.f);
  }
  __syncthreads();
  if (lane < 9) {
    int i = lane;
    float mx = -1e30f;
    for (int j = 0; j < 9; j++) mx = fmaxf(mx, sp[h][i * 9 + j]);
    float e[9], s = 0.f;
    for (int j = 0; j < 9; j++) {
      e[j] = __expf(sp[h][i * 9 + j] - mx);
      s += e[j];
    }
    float inv = 1.f / s;
    for (int j = 0; j < 9; j++) sp[h][i * 9 + j] = e[j] * inv;
  }
  __syncthreads();
  for (int t = lane; t < 576; t += 32) {
    int i = t >> 6, d = t & 63;
    float o = 0.f;
#pragma unroll
    for (int j = 0; j < 9; j++) o += sp[h][i * 9 + j] * sv[h][j * 64 + d];
    out[(size_t)(w * 9 + i) * Dc + h * HDc + d] = (bf16_t)o;
  }
}

// ---------------------------- small utility kernels ------------------------
__global__ void f2bf_kernel(const float* __restrict__ s, bf16_t* __restrict__ d,
                            int n) {
  int t = blockIdx.x * 256 + threadIdx.x;
  if (t < n) d[t] = (bf16_t)s[t];
}

__global__ void pfx_kernel(const float* __restrict__ hs,
                           bf16_t* __restrict__ pfx) {
  int t = blockIdx.x * 256 + threadIdx.x;   // 512*512
  if (t >= 512 * 512) return;
  int d = t & 511, m = t >> 9;
  int b = m >> 4, i = m & 15;
  pfx[t] = (bf16_t)hs[(size_t)b * 916 * Dc + (size_t)i * Dc + d];
}

__global__ void mean_kernel(const float* __restrict__ hs,
                            float* __restrict__ mean) {
  int t = blockIdx.x * 256 + threadIdx.x;   // 32*512
  if (t >= Bc * Dc) return;
  int b = t >> 9, d = t & 511;
  const float* p = hs + (size_t)b * 916 * Dc + d;
  float s = 0.f;
#pragma unroll
  for (int i = 0; i < PFXc; i++) s += p[(size_t)i * Dc];
  mean[t] = s * (1.f / PFXc);
}

// residual = grid + mean(prefix);  xw = bf16(windowed(rolled(residual)))
__global__ void prep_kernel(const float* __restrict__ hs,
                            const float* __restrict__ mean,
                            float* __restrict__ residual,
                            bf16_t* __restrict__ xw) {
  size_t t = (size_t)blockIdx.x * 256 + threadIdx.x;   // 32*900*512
  if (t >= (size_t)Bc * SEQc * Dc) return;
  int d = (int)(t & 511);
  size_t rrow = t >> 9;
  int s = (int)(rrow % SEQc);
  int b = (int)(rrow / SEQc);
  float g = hs[(size_t)b * 916 * Dc + (size_t)(PFXc + s) * Dc + d] +
            mean[b * Dc + d];
  residual[t] = g;
  int h = s / GRIDc, w = s - h * GRIDc;
  int hp = (h + GRIDc - 1) % GRIDc, wp = (w + GRIDc - 1) % GRIDc;
  int row = ((b * 100 + (hp / 3) * 10 + (wp / 3)) * 9 + (hp % 3) * 3 + (wp % 3));
  xw[(size_t)row * Dc + d] = (bf16_t)g;
}

// row-wise RMS norm over 512 cols; one wave per row (wave32, shfl reduce)
__global__ void __launch_bounds__(256)
rms_kernel(const float* __restrict__ src, float* __restrict__ dst,
           bf16_t* __restrict__ dstb, int nrows) {
  int wave = threadIdx.x >> 5, lane = threadIdx.x & 31;
  int row = blockIdx.x * 8 + wave;
  if (row >= nrows) return;
  const float* x = src + (size_t)row * Dc;
  float v[16], ss = 0.f;
#pragma unroll
  for (int j = 0; j < 16; j++) {
    v[j] = x[lane + j * 32];
    ss += v[j] * v[j];
  }
#pragma unroll
  for (int o = 16; o >= 1; o >>= 1) ss += __shfl_xor(ss, o, 32);
  float inv = rsqrtf(ss * (1.f / Dc) + EPSc);
  float* y = dst + (size_t)row * Dc;
#pragma unroll
  for (int j = 0; j < 16; j++) {
    float r = v[j] * inv;
    y[lane + j * 32] = r;
    if (dstb) dstb[(size_t)row * Dc + lane + j * 32] = (bf16_t)r;
  }
}

// ---------------------------------------------------------------------------
extern "C" void kernel_launch(void* const* d_in, const int* in_sizes, int n_in,
                              void* d_out, int out_size, void* d_ws,
                              size_t ws_size, hipStream_t stream) {
  const float* hs      = (const float*)d_in[0];
  const float* qkv_w   = (const float*)d_in[1];
  const float* qkv_b   = (const float*)d_in[2];
  const float* proj_w  = (const float*)d_in[3];
  const float* proj_b  = (const float*)d_in[4];
  const float* rel_b   = (const float*)d_in[5];
  const float* gu_w    = (const float*)d_in[6];
  const float* down_w  = (const float*)d_in[7];
  const float* pgu_w   = (const float*)d_in[8];
  const float* pdown_w = (const float*)d_in[9];
  float* out = (float*)d_out;

  char* p = (char*)d_ws;
  auto alloc = [&](size_t bytes) -> char* {
    char* r = p;
    p += (bytes + 255) & ~(size_t)255;
    return r;
  };
  bf16_t* wq   = (bf16_t*)alloc((size_t)3 * Dc * Dc * 2);           // 1536x512
  bf16_t* wp   = (bf16_t*)alloc((size_t)Dc * Dc * 2);               // 512x512
  bf16_t* wgu  = (bf16_t*)alloc((size_t)2 * INTERc * Dc * 2);       // 3072x512
  bf16_t* wdn  = (bf16_t*)alloc((size_t)Dc * INTERc * 2);           // 512x1536
  bf16_t* wpgu = (bf16_t*)alloc((size_t)2 * INTERc * Dc * 2);
  bf16_t* wpdn = (bf16_t*)alloc((size_t)Dc * INTERc * 2);
  bf16_t* pfxb = (bf16_t*)alloc((size_t)512 * Dc * 2);
  float*  meanb= (float*)alloc((size_t)Bc * Dc * 4);
  bf16_t* phb  = (bf16_t*)alloc((size_t)512 * INTERc * 2);
  float*  resid= (float*)alloc((size_t)Bc * SEQc * Dc * 4);         // 59 MB
  bf16_t* xw   = (bf16_t*)alloc((size_t)NTOK * Dc * 2);             // 29.5 MB
  bf16_t* qkvb = (bf16_t*)alloc((size_t)TOTROWS * INTERc * 2);      // 90 MB (qkv then h)
  bf16_t* attnb= (bf16_t*)alloc((size_t)NTOK * Dc * 2);             // 29.5 MB
  float*  hspre= (float*)alloc((size_t)TOTROWS * Dc * 4);           // 60 MB
  float*  hsmid= (float*)alloc((size_t)TOTROWS * Dc * 4);           // 60 MB
  bf16_t* hsmb = (bf16_t*)alloc((size_t)TOTROWS * Dc * 2);          // 30 MB
  bf16_t* hbuf = qkvb;                       // alias: qkv dead after attention
  float*  fpre = resid;                      // alias: resid+xw dead by then

  // -- weights -> bf16
  auto cvt = [&](const float* s, bf16_t* d, int n) {
    f2bf_kernel<<<(n + 255) / 256, 256, 0, stream>>>(s, d, n);
  };
  cvt(qkv_w,   wq,   3 * Dc * Dc);
  cvt(proj_w,  wp,   Dc * Dc);
  cvt(gu_w,    wgu,  2 * INTERc * Dc);
  cvt(down_w,  wdn,  Dc * INTERc);
  cvt(pgu_w,   wpgu, 2 * INTERc * Dc);
  cvt(pdown_w, wpdn, Dc * INTERc);
  pfx_kernel<<<(512 * 512 + 255) / 256, 256, 0, stream>>>(hs, pfxb);

  // -- prefix mean, grid build, window/roll
  mean_kernel<<<(Bc * Dc + 255) / 256, 256, 0, stream>>>(hs, meanb);
  prep_kernel<<<(Bc * SEQc * Dc) / 256, 256, 0, stream>>>(hs, meanb, resid, xw);

  // -- qkv GEMM (28800 x 1536 x 512)
  gemm_qkv_kernel<<<dim3(NTOK / 128, (3 * Dc) / 64), 256, 0, stream>>>(
      xw, wq, qkv_b, qkvb);

  // -- windowed attention
  attn_kernel<<<NWIN, 256, 0, stream>>>(qkvb, rel_b, attnb);

  // -- proj GEMM + un-window/un-roll + residual (28800 x 512 x 512)
  gemm_proj_kernel<<<dim3(NTOK / 128, Dc / 64), 256, 0, stream>>>(
      attnb, wp, proj_b, resid, hspre);

  // -- prefix swiglu path (512 rows)
  gemm_gu_kernel<<<dim3(512 / 128, INTERc / 64), 256, 0, stream>>>(
      pfxb, wpgu, phb, Dc);
  gemm_down_kernel<<<dim3(512 / 128, Dc / 64), 256, 0, stream>>>(
      phb, wpdn, hs, hspre, INTERc, /*mode=*/1);

  // -- rms over concat(prefix, x) rows -> hs_mid (f32 + bf16)
  rms_kernel<<<(TOTROWS + 7) / 8, 256, 0, stream>>>(hspre, hsmid, hsmb,
                                                    TOTROWS);

  // -- main swiglu: gate/up (29312 x 1536(x2) x 512) then down (x 512 x 1536)
  gemm_gu_kernel<<<dim3(TOTROWS / 128, INTERc / 64), 256, 0, stream>>>(
      hsmb, wgu, hbuf, Dc);
  gemm_down_kernel<<<dim3(TOTROWS / 128, Dc / 64), 256, 0, stream>>>(
      hbuf, wdn, hsmid, fpre, INTERc, /*mode=*/0);

  // -- final rms -> d_out
  rms_kernel<<<(TOTROWS + 7) / 8, 256, 0, stream>>>(fpre, out, nullptr,
                                                    TOTROWS);
}
